// egc_63754494542122
// MI455X (gfx1250) — matrix operand order repaired and checked by reading, not compiled, not run on hardware
//
#include <hip/hip_runtime.h>

// ---------------------------------------------------------------------------
// EGNN layer for MI455X (gfx1250, wave32, WMMA).
//
// Orientation: features on M/K, edges/nodes on N (16 per tile).
//   A = transposed f16 weights (global, L2-resident), ISA A-frag layout
//   B = per-edge/node activations in LDS row-major -> contiguous B-frag loads
//       (compiler hoists them into registers across the mt loop)
//   C = f32 accum; packs 8 consecutive features/lane -> one ds_store_b128
// Each wave processes NT=2 node-tiles so every A-fragment load feeds two
// WMMAs (doubles FLOP per L2 byte). One wave per block: no cross-wave sync
// is needed anywhere (segment_sum == intra-tile reduction since the 16
// in-edges of node n are edges 16n..16n+15).
// tanh -> native v_tanh_f32 (gfx1250 TRANS op, co-executes with WMMA).
// ---------------------------------------------------------------------------

typedef _Float16 half8  __attribute__((ext_vector_type(8)));
typedef _Float16 half16 __attribute__((ext_vector_type(16)));
typedef float    v8f    __attribute__((ext_vector_type(8)));

#define KP1 288          // padded K for [h_src(128) | h_dst(128) | l2 | 0...]
// f16 weight table offsets in d_ws (in halves)
#define OFF_W1T   0       // [128][288]
#define OFF_W2T   36864   // [128][128]
#define OFF_WC1T  53248   // [128][128]
#define OFF_WH1T  69632   // [128][256]
#define OFF_WH2T  102400  // [128][128]
#define HALVES_TOTAL 118784
#define MI_BYTE_OFF  237568  // 16B-aligned start of m_i f32 [N][128]

static __device__ __forceinline__ float fast_tanh(float x) {
#if __has_builtin(__builtin_amdgcn_tanhf)
  return __builtin_amdgcn_tanhf(x);        // v_tanh_f32
#elif __has_builtin(__builtin_amdgcn_tanh_f32)
  return __builtin_amdgcn_tanh_f32(x);
#else
  return tanhf(x);
#endif
}

static __device__ __forceinline__ half16 h16cat(half8 lo, half8 hi) {
  half16 r;
#pragma unroll
  for (int i = 0; i < 8; ++i) { r[i] = lo[i]; r[i + 8] = hi[i]; }
  return r;
}

// A fragment (16x32 f16) from a transposed weight table W[mrows][kp]:
// lane holds row M = mt*16 + (lane&15); lane-half 'hi' picks K chunks
// {k0+8hi..+7} and {k0+16+8hi..+7} per the ISA 16-bit A layout.
static __device__ __forceinline__ half16 load_a(const _Float16* W, int kp,
                                                int row, int k0, int hi) {
  const _Float16* r = W + (size_t)row * kp + k0 + hi * 8;
  return h16cat(*(const half8*)r, *(const half8*)(r + 16));
}

// B fragment (32x16 f16) from LDS row (one edge/node's feature row):
// lane = column N, lane-half 'hi' picks K chunks {k0+8hi..+7} and
// {k0+16+8hi..+7} (same K order as the A fragment).
static __device__ __forceinline__ half16 load_b(const _Float16* row, int k0,
                                                int hi) {
  const _Float16* p = row + k0 + hi * 8;
  return h16cat(*(const half8*)p, *(const half8*)(p + 16));
}

// ---------------------------------------------------------------------------
// Weight prep: transpose + f32->f16 all five matrices into d_ws.
// W1 rows reordered: k 0..255 <- W1 rows 1..256 (src|dst), k=256 <- row 0
// (the l2 weight), k 257..287 zero-padded.
// ---------------------------------------------------------------------------
__global__ void egc_prep_weights(const float* __restrict__ W1,
                                 const float* __restrict__ W2,
                                 const float* __restrict__ Wc1,
                                 const float* __restrict__ Wh1,
                                 const float* __restrict__ Wh2,
                                 _Float16* __restrict__ ws) {
  int i = blockIdx.x * blockDim.x + threadIdx.x;
  if (i >= HALVES_TOTAL) return;
  float v = 0.f;
  if (i < OFF_W2T) {                       // W1T [128][288]
    int m = i / KP1, k = i % KP1;
    if (k < 256)       v = W1[(size_t)(k + 1) * 128 + m];
    else if (k == 256) v = W1[m];
  } else if (i < OFF_WC1T) {               // W2T [128][128]
    int r = i - OFF_W2T; int m = r / 128, k = r % 128;
    v = W2[(size_t)k * 128 + m];
  } else if (i < OFF_WH1T) {               // Wc1T [128][128]
    int r = i - OFF_WC1T; int m = r / 128, k = r % 128;
    v = Wc1[(size_t)k * 128 + m];
  } else if (i < OFF_WH2T) {               // Wh1T [128][256]
    int r = i - OFF_WH1T; int m = r / 256, k = r % 256;
    v = Wh1[(size_t)k * 128 + m];
  } else {                                 // Wh2T [128][128]
    int r = i - OFF_WH2T; int m = r / 128, k = r % 128;
    v = Wh2[(size_t)k * 128 + m];
  }
  ws[i] = (_Float16)v;
}

// ---------------------------------------------------------------------------
// Edge kernel: one wave (== one block) handles NT=2 nodes == 2 tiles of 16
// contiguous in-edges. Chain per tile:
//   x(16x288) -> tanh GEMM1 -> t -> GEMM2 -> m -> tanh GEMM3 -> u
//   -> w = u . Wc2 ; outputs: m_i (ws), coords_out.
// 272 WMMAs per wave (two per A-fragment load).
// ---------------------------------------------------------------------------
__global__ __launch_bounds__(32) void egc_edge_kernel(
    const float* __restrict__ coords, const float* __restrict__ hidden,
    const int* __restrict__ edges, const float* __restrict__ b1,
    const float* __restrict__ b2, const float* __restrict__ bc1,
    const float* __restrict__ Wc2, const _Float16* __restrict__ wsW,
    float* __restrict__ mi_out, float* __restrict__ coords_out, int N, int E) {
  __shared__ __attribute__((aligned(16))) _Float16 XB[2][16][KP1];
  __shared__ __attribute__((aligned(16))) _Float16 TB[2][16][128];
  __shared__ __attribute__((aligned(16))) _Float16 MB[2][16][128];
  __shared__ __attribute__((aligned(16))) float    CT[2][16][4];

  const int lane = threadIdx.x & 31;
  const int j = lane & 15, hi = lane >> 4;
  const int nb = blockIdx.x * 2;
  if (nb >= N) return;                      // wave-uniform; EXEC stays all-1s

  __builtin_prefetch(wsW, 0, 3);            // global_prefetch_b8 (weights)

  const int* srcA = edges;
  const int* dstA = edges + E;

  float dxs[2], dys[2], dzs[2];
  int nn[2];
#pragma unroll
  for (int t = 0; t < 2; ++t) {
    const int n = min(nb + t, N - 1);       // clamp (N even -> always valid)
    nn[t] = n;
    const int e = n * 16 + j;               // DEG == 16 (matches WMMA N)
    const int s = srcA[e], d = dstA[e];     // d == n for this edge structure
    const float dx = coords[3 * s + 0] - coords[3 * d + 0];
    const float dy = coords[3 * s + 1] - coords[3 * d + 1];
    const float dz = coords[3 * s + 2] - coords[3 * d + 2];
    dxs[t] = dx; dys[t] = dy; dzs[t] = dz;
    const float l2 = sqrtf(dx * dx + dy * dy + dz * dz);

    // Stage x row-major per edge: lanes 0-15 -> h[src] (cols 0..127),
    // lanes 16-31 -> h[dst] (cols 128..255); l2 at col 256; pad zeros.
    const float* hrow = hidden + (size_t)(hi ? d : s) * 128;
#pragma unroll
    for (int c = 0; c < 128; c += 8) {
      half8 h;
#pragma unroll
      for (int q = 0; q < 8; ++q) h[q] = (_Float16)hrow[c + q];
      *(half8*)&XB[t][j][hi * 128 + c] = h;
    }
    if (hi == 0) {
      XB[t][j][256] = (_Float16)l2;
#pragma unroll
      for (int c = 257; c < KP1; ++c) XB[t][j][c] = (_Float16)0.f;
    }
  }

  const _Float16* W1T  = wsW + OFF_W1T;
  const _Float16* W2T  = wsW + OFF_W2T;
  const _Float16* Wc1T = wsW + OFF_WC1T;

  // GEMM1: t = tanh(W1T(128x288) . xT + b1) -> TB   (2 x 72 WMMA)
#pragma unroll 1
  for (int mt = 0; mt < 8; ++mt) {
    v8f a0, a1;
#pragma unroll
    for (int g = 0; g < 8; ++g) { float bv = b1[mt * 16 + hi * 8 + g]; a0[g] = bv; a1[g] = bv; }
#pragma unroll
    for (int k = 0; k < KP1; k += 32) {
      half16 a  = load_a(W1T, KP1, mt * 16 + j, k, hi);
      half16 f0 = load_b(&XB[0][j][0], k, hi);
      half16 f1 = load_b(&XB[1][j][0], k, hi);
      a0 = __builtin_amdgcn_wmma_f32_16x16x32_f16(false, a, false, f0, (short)0, a0, false, false);
      a1 = __builtin_amdgcn_wmma_f32_16x16x32_f16(false, a, false, f1, (short)0, a1, false, false);
    }
    half8 t0, t1;
#pragma unroll
    for (int g = 0; g < 8; ++g) { t0[g] = (_Float16)fast_tanh(a0[g]); t1[g] = (_Float16)fast_tanh(a1[g]); }
    *(half8*)&TB[0][j][mt * 16 + hi * 8] = t0;   // 8 consecutive feats/lane
    *(half8*)&TB[1][j][mt * 16 + hi * 8] = t1;
  }

  // GEMM2: m = W2T . tT + b2 (linear) -> MB   (2 x 32 WMMA)
#pragma unroll 1
  for (int mt = 0; mt < 8; ++mt) {
    v8f a0, a1;
#pragma unroll
    for (int g = 0; g < 8; ++g) { float bv = b2[mt * 16 + hi * 8 + g]; a0[g] = bv; a1[g] = bv; }
#pragma unroll
    for (int k = 0; k < 128; k += 32) {
      half16 a  = load_a(W2T, 128, mt * 16 + j, k, hi);
      half16 f0 = load_b(&TB[0][j][0], k, hi);
      half16 f1 = load_b(&TB[1][j][0], k, hi);
      a0 = __builtin_amdgcn_wmma_f32_16x16x32_f16(false, a, false, f0, (short)0, a0, false, false);
      a1 = __builtin_amdgcn_wmma_f32_16x16x32_f16(false, a, false, f1, (short)0, a1, false, false);
    }
    half8 t0, t1;
#pragma unroll
    for (int g = 0; g < 8; ++g) { t0[g] = (_Float16)a0[g]; t1[g] = (_Float16)a1[g]; }
    *(half8*)&MB[0][j][mt * 16 + hi * 8] = t0;
    *(half8*)&MB[1][j][mt * 16 + hi * 8] = t1;
  }

  // GEMM3: u = tanh(Wc1T . mT + bc1) -> TB (reuse)   (2 x 32 WMMA)
#pragma unroll 1
  for (int mt = 0; mt < 8; ++mt) {
    v8f a0, a1;
#pragma unroll
    for (int g = 0; g < 8; ++g) { float bv = bc1[mt * 16 + hi * 8 + g]; a0[g] = bv; a1[g] = bv; }
#pragma unroll
    for (int k = 0; k < 128; k += 32) {
      half16 a  = load_a(Wc1T, 128, mt * 16 + j, k, hi);
      half16 f0 = load_b(&MB[0][j][0], k, hi);
      half16 f1 = load_b(&MB[1][j][0], k, hi);
      a0 = __builtin_amdgcn_wmma_f32_16x16x32_f16(false, a, false, f0, (short)0, a0, false, false);
      a1 = __builtin_amdgcn_wmma_f32_16x16x32_f16(false, a, false, f1, (short)0, a1, false, false);
    }
    half8 t0, t1;
#pragma unroll
    for (int g = 0; g < 8; ++g) { t0[g] = (_Float16)fast_tanh(a0[g]); t1[g] = (_Float16)fast_tanh(a1[g]); }
    *(half8*)&TB[0][j][mt * 16 + hi * 8] = t0;
    *(half8*)&TB[1][j][mt * 16 + hi * 8] = t1;
  }

#pragma unroll
  for (int t = 0; t < 2; ++t) {
    // Coord head: w[e] = u[e] . Wc2 (two lanes per edge, 64 feats each)
    float wsum = 0.f;
#pragma unroll
    for (int c = 0; c < 64; ++c)
      wsum += (float)TB[t][j][hi * 64 + c] * Wc2[hi * 64 + c];
    wsum += __shfl_xor(wsum, 16, 32);

    if (hi == 0) {
      CT[t][j][0] = dxs[t] * wsum;
      CT[t][j][1] = dys[t] * wsum;
      CT[t][j][2] = dzs[t] * wsum;
    }
    const bool valid = (nb + t < N);
    // coords_out[n] = coords[n] + (sum_e ct)/16   (intra-tile segment_sum)
    if (valid && lane < 3) {
      float ssum = 0.f;
#pragma unroll
      for (int q = 0; q < 16; ++q) ssum += CT[t][q][lane];
      coords_out[3 * nn[t] + lane] = coords[3 * nn[t] + lane] + ssum * (1.0f / 16.0f);
    }
    // m_i[n] = sum over 16 edges of m   (each lane owns 4 feature columns)
    if (valid) {
      const int f = lane * 4;
      float s0 = 0.f, s1 = 0.f, s2 = 0.f, s3 = 0.f;
#pragma unroll
      for (int q = 0; q < 16; ++q) {
        s0 += (float)MB[t][q][f + 0]; s1 += (float)MB[t][q][f + 1];
        s2 += (float)MB[t][q][f + 2]; s3 += (float)MB[t][q][f + 3];
      }
      float* mo = mi_out + (size_t)nn[t] * 128 + f;
      mo[0] = s0; mo[1] = s1; mo[2] = s2; mo[3] = s3;
    }
  }
}

// ---------------------------------------------------------------------------
// Node kernel: one wave handles 2 tiles of 16 nodes. hmi = [hidden | m_i],
// h = tanh(hmi.Wh1+bh1).Wh2+bh2 ; hidden_out = hidden + h. 192 WMMAs/wave.
// ---------------------------------------------------------------------------
__global__ __launch_bounds__(32) void egc_node_kernel(
    const float* __restrict__ hidden, const float* __restrict__ bh1,
    const float* __restrict__ bh2, const _Float16* __restrict__ wsW,
    const float* __restrict__ mi, float* __restrict__ hidden_out, int N) {
  __shared__ __attribute__((aligned(16))) _Float16 HB[2][16][256];
  __shared__ __attribute__((aligned(16))) _Float16 TB[2][16][128];

  const int lane = threadIdx.x & 31;
  const int j = lane & 15, hi = lane >> 4;
  const int ntiles = (N + 15) / 16;
  const int tb0 = blockIdx.x * 2;
  if (tb0 >= ntiles) return;

  __builtin_prefetch(wsW + OFF_WH1T, 0, 3);

  int node_t[2];
#pragma unroll
  for (int t = 0; t < 2; ++t) {
    const int tile = min(tb0 + t, ntiles - 1);
    const int node = min(tile * 16 + j, N - 1);
    node_t[t] = node;
    // Stage hmi: lanes 0-15 -> hidden row (cols 0..127), 16-31 -> m_i row.
    const float* row = hi ? (mi + (size_t)node * 128)
                          : (hidden + (size_t)node * 128);
#pragma unroll
    for (int c = 0; c < 128; c += 8) {
      half8 h;
#pragma unroll
      for (int q = 0; q < 8; ++q) h[q] = (_Float16)row[c + q];
      *(half8*)&HB[t][j][hi * 128 + c] = h;
    }
  }

  const _Float16* Wh1T = wsW + OFF_WH1T;   // [128][256]
  const _Float16* Wh2T = wsW + OFF_WH2T;   // [128][128]

  // GEMM4: t = tanh(Wh1T . hmiT + bh1) -> TB   (2 x 64 WMMA)
#pragma unroll 1
  for (int mt = 0; mt < 8; ++mt) {
    v8f a0, a1;
#pragma unroll
    for (int g = 0; g < 8; ++g) { float bv = bh1[mt * 16 + hi * 8 + g]; a0[g] = bv; a1[g] = bv; }
#pragma unroll
    for (int k = 0; k < 256; k += 32) {
      half16 a  = load_a(Wh1T, 256, mt * 16 + j, k, hi);
      half16 f0 = load_b(&HB[0][j][0], k, hi);
      half16 f1 = load_b(&HB[1][j][0], k, hi);
      a0 = __builtin_amdgcn_wmma_f32_16x16x32_f16(false, a, false, f0, (short)0, a0, false, false);
      a1 = __builtin_amdgcn_wmma_f32_16x16x32_f16(false, a, false, f1, (short)0, a1, false, false);
    }
    half8 t0, t1;
#pragma unroll
    for (int g = 0; g < 8; ++g) { t0[g] = (_Float16)fast_tanh(a0[g]); t1[g] = (_Float16)fast_tanh(a1[g]); }
    *(half8*)&TB[0][j][mt * 16 + hi * 8] = t0;
    *(half8*)&TB[1][j][mt * 16 + hi * 8] = t1;
  }

  // GEMM5: h = Wh2T . tT + bh2 ; hidden_out = hidden + h   (2 x 32 WMMA)
#pragma unroll 1
  for (int mt = 0; mt < 8; ++mt) {
    v8f a0, a1;
#pragma unroll
    for (int g = 0; g < 8; ++g) { float bv = bh2[mt * 16 + hi * 8 + g]; a0[g] = bv; a1[g] = bv; }
#pragma unroll
    for (int k = 0; k < 128; k += 32) {
      half16 a  = load_a(Wh2T, 128, mt * 16 + j, k, hi);
      half16 f0 = load_b(&TB[0][j][0], k, hi);
      half16 f1 = load_b(&TB[1][j][0], k, hi);
      a0 = __builtin_amdgcn_wmma_f32_16x16x32_f16(false, a, false, f0, (short)0, a0, false, false);
      a1 = __builtin_amdgcn_wmma_f32_16x16x32_f16(false, a, false, f1, (short)0, a1, false, false);
    }
    // C layout: lane's column = node j, rows = 8 consecutive output feats.
#pragma unroll
    for (int t = 0; t < 2; ++t) {
      const bool valid = (tb0 + t < ntiles) && ((tb0 + t) * 16 + j < N);
      if (valid) {
        const int node = node_t[t];
        const float* hv = hidden + (size_t)node * 128 + mt * 16 + hi * 8;
        float* ov = hidden_out + (size_t)node * 128 + mt * 16 + hi * 8;
        const v8f& av = (t == 0) ? a0 : a1;
#pragma unroll
        for (int g = 0; g < 8; ++g) ov[g] = hv[g] + av[g];
      }
    }
  }
}

// ---------------------------------------------------------------------------
extern "C" void kernel_launch(void* const* d_in, const int* in_sizes, int n_in,
                              void* d_out, int out_size, void* d_ws,
                              size_t ws_size, hipStream_t stream) {
  const float* coords = (const float*)d_in[0];
  const float* hidden = (const float*)d_in[1];
  const int*   edges  = (const int*)d_in[2];
  const float* W1  = (const float*)d_in[3];
  const float* b1  = (const float*)d_in[4];
  const float* W2  = (const float*)d_in[5];
  const float* b2  = (const float*)d_in[6];
  const float* Wc1 = (const float*)d_in[7];
  const float* bc1 = (const float*)d_in[8];
  const float* Wc2 = (const float*)d_in[9];
  const float* Wh1 = (const float*)d_in[10];
  const float* bh1 = (const float*)d_in[11];
  const float* Wh2 = (const float*)d_in[12];
  const float* bh2 = (const float*)d_in[13];
  (void)n_in; (void)out_size; (void)ws_size;

  const int N = in_sizes[0] / 3;
  const int E = in_sizes[2] / 2;

  _Float16* wsH = (_Float16*)d_ws;
  float* mi = (float*)((char*)d_ws + MI_BYTE_OFF);
  float* coords_out = (float*)d_out;
  float* hidden_out = coords_out + (size_t)N * 3;

  egc_prep_weights<<<(HALVES_TOTAL + 255) / 256, 256, 0, stream>>>(
      W1, W2, Wc1, Wh1, Wh2, wsH);
  egc_edge_kernel<<<(N + 1) / 2, 32, 0, stream>>>(
      coords, hidden, edges, b1, b2, bc1, Wc2, wsH, mi, coords_out, N, E);
  const int ntiles = (N + 15) / 16;
  egc_node_kernel<<<(ntiles + 1) / 2, 32, 0, stream>>>(
      hidden, bh1, bh2, wsH, mi, hidden_out, N);
}
